// PEER_1726576855273
// MI455X (gfx1250) — matrix operand order repaired and checked by reading, not compiled, not run on hardware
//
#include <hip/hip_runtime.h>
#include <math.h>

typedef __bf16 bf16;
typedef __attribute__((ext_vector_type(4)))  __bf16 v4bf;
typedef __attribute__((ext_vector_type(8)))  __bf16 v8bf;
typedef __attribute__((ext_vector_type(16))) __bf16 v16bf;
typedef __attribute__((ext_vector_type(8)))  float  v8f;

#define DIM    1024
#define HEADS  8
#define NKEYS  256
#define DKEY   512
#define TOK    8192      // BATCH * SEQ
#define NCOL   4096      // 2 * HEADS * NKEYS
#define WQCOLS 8192      // 2 * HEADS * DKEY

#if defined(__gfx1250__) && __has_builtin(__builtin_amdgcn_global_load_async_to_lds_b128)
#define HAVE_ASYNC_LDS 1
typedef __attribute__((ext_vector_type(4))) int v4i;
typedef __attribute__((address_space(1))) v4i as1_v4i;   // global 16B packet
typedef __attribute__((address_space(3))) v4i as3_v4i;   // LDS 16B packet
#else
#define HAVE_ASYNC_LDS 0
#endif

#if HAVE_ASYNC_LDS
static __device__ __forceinline__ void async_wait4() {
#if __has_builtin(__builtin_amdgcn_s_wait_asynccnt)
  __builtin_amdgcn_s_wait_asynccnt(4);
#else
  asm volatile("s_wait_asynccnt 0x4" ::: "memory");
#endif
}
static __device__ __forceinline__ void async_wait0() {
#if __has_builtin(__builtin_amdgcn_s_wait_asynccnt)
  __builtin_amdgcn_s_wait_asynccnt(0);
#else
  asm volatile("s_wait_asynccnt 0x0" ::: "memory");
#endif
}
#endif

static __device__ __forceinline__ v16bf cat8(v8bf lo, v8bf hi) {
  return __builtin_shufflevector(lo, hi, 0,1,2,3,4,5,6,7,8,9,10,11,12,13,14,15);
}

// ---------------------------------------------------------------- f32 -> bf16
__global__ void __launch_bounds__(256)
cvt_f32_bf16(const float* __restrict__ s, bf16* __restrict__ d, int n) {
  int i = (blockIdx.x * blockDim.x + threadIdx.x) * 4;
  if (i >= n) return;
  float4 v = *(const float4*)(s + i);
  v4bf o;
  o[0] = (bf16)v.x; o[1] = (bf16)v.y; o[2] = (bf16)v.z; o[3] = (bf16)v.w;
  *(v4bf*)(d + i) = o;
}

// ------------------------------------------------- W2[pg] = Wq_slice @ K[pg]^T
// M = DIM (d), N = NKEYS (key idx), K = DKEY.  Output stored transposed (W2T)
// as bf16 so the sim GEMM's B-fragments are contiguous row reads.
__global__ void __launch_bounds__(128)
gemm_w2(const bf16* __restrict__ wq, const bf16* __restrict__ keys,
        bf16* __restrict__ w2t) {
  __shared__ __align__(16) bf16 As[2][64][40];
  __shared__ __align__(16) bf16 Bs[2][64][40];
  const int pg = blockIdx.z, p = pg >> 3, h = pg & 7;
  const int gm0 = blockIdx.x * 64;          // d
  const int gn0 = blockIdx.y * 64;          // key index
  const int lane = threadIdx.x & 31, wave = threadIdx.x >> 5;
  const int l16 = lane & 15, half = lane >> 4;
  const int wm = (wave >> 1) * 32, wn = (wave & 1) * 32;
  const int r0 = threadIdx.x >> 2;          // stage-in row (and row+32)
  const int cc = (threadIdx.x & 3) * 8;     // stage-in column chunk
  v8f acc[2][2] = {};

  auto stage = [&](int buf, int kb) {
    const bf16* ga0 = wq + (size_t)(gm0 + r0) * WQCOLS + pg * DKEY + kb + cc;
    const bf16* ga1 = wq + (size_t)(gm0 + r0 + 32) * WQCOLS + pg * DKEY + kb + cc;
    const bf16* gb0 = keys + ((size_t)(h * NKEYS + gn0 + r0) * 2 + p) * DKEY + kb + cc;
    const bf16* gb1 = keys + ((size_t)(h * NKEYS + gn0 + r0 + 32) * 2 + p) * DKEY + kb + cc;
#if HAVE_ASYNC_LDS
    __builtin_amdgcn_global_load_async_to_lds_b128(
        (as1_v4i*)ga0, (as3_v4i*)&As[buf][r0][cc], 0, 0);
    __builtin_amdgcn_global_load_async_to_lds_b128(
        (as1_v4i*)ga1, (as3_v4i*)&As[buf][r0 + 32][cc], 0, 0);
    __builtin_amdgcn_global_load_async_to_lds_b128(
        (as1_v4i*)gb0, (as3_v4i*)&Bs[buf][r0][cc], 0, 0);
    __builtin_amdgcn_global_load_async_to_lds_b128(
        (as1_v4i*)gb1, (as3_v4i*)&Bs[buf][r0 + 32][cc], 0, 0);
#else
    *(v8bf*)&As[buf][r0][cc]      = *(const v8bf*)ga0;
    *(v8bf*)&As[buf][r0 + 32][cc] = *(const v8bf*)ga1;
    *(v8bf*)&Bs[buf][r0][cc]      = *(const v8bf*)gb0;
    *(v8bf*)&Bs[buf][r0 + 32][cc] = *(const v8bf*)gb1;
#endif
  };

  constexpr int STEPS = DKEY / 32;
  stage(0, 0);
  for (int i = 0; i < STEPS; ++i) {
    if (i + 1 < STEPS) stage((i + 1) & 1, (i + 1) * 32);
#if HAVE_ASYNC_LDS
    if (i + 1 < STEPS) async_wait4(); else async_wait0();
#endif
    __syncthreads();
    const int buf = i & 1;
    v16bf af[2], bfr[2];
#pragma unroll
    for (int ti = 0; ti < 2; ++ti) {
      v8bf lo = *(const v8bf*)&As[buf][wm + 16 * ti + l16][8 * half];
      v8bf hi = *(const v8bf*)&As[buf][wm + 16 * ti + l16][16 + 8 * half];
      af[ti] = cat8(lo, hi);
    }
#pragma unroll
    for (int tj = 0; tj < 2; ++tj) {
      v8bf lo = *(const v8bf*)&Bs[buf][wn + 16 * tj + l16][8 * half];
      v8bf hi = *(const v8bf*)&Bs[buf][wn + 16 * tj + l16][16 + 8 * half];
      bfr[tj] = cat8(lo, hi);
    }
#pragma unroll
    for (int ti = 0; ti < 2; ++ti)
#pragma unroll
      for (int tj = 0; tj < 2; ++tj)
        acc[ti][tj] = __builtin_amdgcn_wmma_f32_16x16x32_bf16(
            false, af[ti], false, bfr[tj], (short)0, acc[ti][tj], false, false);
    __syncthreads();
  }
  // store C^T as bf16: lane owns 8 contiguous m for fixed n
#pragma unroll
  for (int ti = 0; ti < 2; ++ti)
#pragma unroll
    for (int tj = 0; tj < 2; ++tj) {
      int n = gn0 + wn + 16 * tj + l16;
      int m = gm0 + wm + 16 * ti + 8 * half;
      v8bf o;
#pragma unroll
      for (int r = 0; r < 8; ++r) o[r] = (bf16)acc[ti][tj][r];
      *(v8bf*)(w2t + (size_t)(pg * NKEYS + n) * DIM + m) = o;
    }
}

// --------------------------------------------------------- sim = x_bf16 @ W2
// M = TOK, N = NCOL, K = DIM.  B supplied transposed (W2T, N-major).
__global__ void __launch_bounds__(128)
gemm_sim(const bf16* __restrict__ xb, const bf16* __restrict__ w2t,
         float* __restrict__ sim) {
  __shared__ __align__(16) bf16 As[2][64][40];
  __shared__ __align__(16) bf16 Bs[2][64][40];
  const int gm0 = blockIdx.x * 64;          // token
  const int gn0 = blockIdx.y * 64;          // sim column
  const int lane = threadIdx.x & 31, wave = threadIdx.x >> 5;
  const int l16 = lane & 15, half = lane >> 4;
  const int wm = (wave >> 1) * 32, wn = (wave & 1) * 32;
  const int r0 = threadIdx.x >> 2;
  const int cc = (threadIdx.x & 3) * 8;
  v8f acc[2][2] = {};

  auto stage = [&](int buf, int kb) {
    const bf16* ga0 = xb + (size_t)(gm0 + r0) * DIM + kb + cc;
    const bf16* ga1 = xb + (size_t)(gm0 + r0 + 32) * DIM + kb + cc;
    const bf16* gb0 = w2t + (size_t)(gn0 + r0) * DIM + kb + cc;
    const bf16* gb1 = w2t + (size_t)(gn0 + r0 + 32) * DIM + kb + cc;
#if HAVE_ASYNC_LDS
    __builtin_amdgcn_global_load_async_to_lds_b128(
        (as1_v4i*)ga0, (as3_v4i*)&As[buf][r0][cc], 0, 0);
    __builtin_amdgcn_global_load_async_to_lds_b128(
        (as1_v4i*)ga1, (as3_v4i*)&As[buf][r0 + 32][cc], 0, 0);
    __builtin_amdgcn_global_load_async_to_lds_b128(
        (as1_v4i*)gb0, (as3_v4i*)&Bs[buf][r0][cc], 0, 0);
    __builtin_amdgcn_global_load_async_to_lds_b128(
        (as1_v4i*)gb1, (as3_v4i*)&Bs[buf][r0 + 32][cc], 0, 0);
#else
    *(v8bf*)&As[buf][r0][cc]      = *(const v8bf*)ga0;
    *(v8bf*)&As[buf][r0 + 32][cc] = *(const v8bf*)ga1;
    *(v8bf*)&Bs[buf][r0][cc]      = *(const v8bf*)gb0;
    *(v8bf*)&Bs[buf][r0 + 32][cc] = *(const v8bf*)gb1;
#endif
  };

  constexpr int STEPS = DIM / 32;
  stage(0, 0);
  for (int i = 0; i < STEPS; ++i) {
    if (i + 1 < STEPS) stage((i + 1) & 1, (i + 1) * 32);
#if HAVE_ASYNC_LDS
    if (i + 1 < STEPS) async_wait4(); else async_wait0();
#endif
    __syncthreads();
    const int buf = i & 1;
    v16bf af[2], bfr[2];
#pragma unroll
    for (int ti = 0; ti < 2; ++ti) {
      v8bf lo = *(const v8bf*)&As[buf][wm + 16 * ti + l16][8 * half];
      v8bf hi = *(const v8bf*)&As[buf][wm + 16 * ti + l16][16 + 8 * half];
      af[ti] = cat8(lo, hi);
    }
#pragma unroll
    for (int tj = 0; tj < 2; ++tj) {
      v8bf lo = *(const v8bf*)&Bs[buf][wn + 16 * tj + l16][8 * half];
      v8bf hi = *(const v8bf*)&Bs[buf][wn + 16 * tj + l16][16 + 8 * half];
      bfr[tj] = cat8(lo, hi);
    }
#pragma unroll
    for (int ti = 0; ti < 2; ++ti)
#pragma unroll
      for (int tj = 0; tj < 2; ++tj)
        acc[ti][tj] = __builtin_amdgcn_wmma_f32_16x16x32_bf16(
            false, af[ti], false, bfr[tj], (short)0, acc[ti][tj], false, false);
    __syncthreads();
  }
#pragma unroll
  for (int ti = 0; ti < 2; ++ti)
#pragma unroll
    for (int tj = 0; tj < 2; ++tj) {
      int m = gm0 + wm + 16 * ti + 8 * half;
      int n = gn0 + wn + 16 * tj + l16;
#pragma unroll
      for (int r = 0; r < 8; ++r)
        sim[(size_t)(m + r) * NCOL + n] = acc[ti][tj][r];
    }
}

// ---------------------- top-1 of outer-sum == argmax(p=0) + argmax(p=1)
__global__ void __launch_bounds__(256)
argmax_kernel(const float* __restrict__ sim, int* __restrict__ eidx,
              float* __restrict__ esc) {
  const int gw = blockIdx.x * 8 + (threadIdx.x >> 5);   // (token, head) wave
  const int lane = threadIdx.x & 31;
  const int tkn = gw >> 3, h = gw & 7;
  float bv[2]; int bi[2];
#pragma unroll
  for (int p = 0; p < 2; ++p) {
    const float* row = sim + (size_t)tkn * NCOL + (p * HEADS + h) * NKEYS;
    float best = -INFINITY; int idx = 0;
#pragma unroll
    for (int j = 0; j < 8; ++j) {
      float v = row[lane + 32 * j];
      if (v > best) { best = v; idx = lane + 32 * j; }
    }
    for (int off = 16; off; off >>= 1) {
      float ov = __shfl_xor(best, off);
      int   oi = __shfl_xor(idx, off);
      if (ov > best || (ov == best && oi < idx)) { best = ov; idx = oi; }
    }
    bv[p] = best; bi[p] = idx;
  }
  if (lane == 0) {
    eidx[gw] = bi[0] * NKEYS + bi[1];
    esc[gw]  = bv[0] + bv[1];
  }
}

// -------- gather + dot + gelu*relu(score) + weighted accumulate (mem-bound)
__global__ void __launch_bounds__(256)
peer_out_kernel(const float* __restrict__ x, const float* __restrict__ wd,
                const float* __restrict__ wu, const int* __restrict__ eidx,
                const float* __restrict__ esc, float* __restrict__ out) {
  const int t = blockIdx.x, tid = threadIdx.x;
  __shared__ float red[8];
  __shared__ float hbc;
  const float4 xv = *(const float4*)(x + (size_t)t * DIM + tid * 4);
  float4 acc = make_float4(0.f, 0.f, 0.f, 0.f);
  for (int h = 0; h < HEADS; ++h) {
    const int   e = eidx[t * HEADS + h];
    const float s = esc[t * HEADS + h];
    const float* wur = wu + (size_t)e * DIM + tid * 4;
    __builtin_prefetch(wur, 0, 1);                     // global_prefetch for w_up row
    const float4 dv = *(const float4*)(wd + (size_t)e * DIM + tid * 4);
    float pr = xv.x * dv.x + xv.y * dv.y + xv.z * dv.z + xv.w * dv.w;
    for (int off = 16; off; off >>= 1) pr += __shfl_xor(pr, off);
    if ((tid & 31) == 0) red[tid >> 5] = pr;
    __syncthreads();
    if (tid == 0) {
      float sum = 0.f;
#pragma unroll
      for (int i = 0; i < 8; ++i) sum += red[i];
      float g = 0.5f * sum * (1.0f + erff(sum * 0.70710678118654752f)); // exact gelu
      hbc = g * fmaxf(s, 0.0f);
    }
    __syncthreads();
    const float g = hbc;
    const float4 uv = *(const float4*)wur;
    acc.x += g * uv.x; acc.y += g * uv.y; acc.z += g * uv.z; acc.w += g * uv.w;
  }
  *(float4*)(out + (size_t)t * DIM + tid * 4) = acc;
}

extern "C" void kernel_launch(void* const* d_in, const int* in_sizes, int n_in,
                              void* d_out, int out_size, void* d_ws, size_t ws_size,
                              hipStream_t stream) {
  const float* x  = (const float*)d_in[0];   // (TOK, DIM)
  const float* wq = (const float*)d_in[1];   // (DIM, WQCOLS)
  const float* ky = (const float*)d_in[2];   // (HEADS, NKEYS, 2, DKEY)
  const float* wd = (const float*)d_in[3];   // (65536, DIM)
  const float* wu = (const float*)d_in[4];   // (65536, DIM)
  float* out = (float*)d_out;

  char* base = (char*)d_ws;
  size_t off = 0;
  auto take = [&](size_t bytes) -> char* {
    char* r = base + off;
    off = (off + bytes + 255) & ~(size_t)255;
    return r;
  };
  bf16*  x_bf  = (bf16*) take((size_t)TOK * DIM * 2);
  bf16*  wq_bf = (bf16*) take((size_t)DIM * WQCOLS * 2);
  bf16*  ky_bf = (bf16*) take((size_t)HEADS * NKEYS * 2 * DKEY * 2);
  bf16*  w2t   = (bf16*) take((size_t)NCOL * DIM * 2);
  float* sim   = (float*)take((size_t)TOK * NCOL * 4);
  int*   eidx  = (int*)  take((size_t)TOK * HEADS * 4);
  float* esc   = (float*)take((size_t)TOK * HEADS * 4);

  cvt_f32_bf16<<<(TOK * DIM / 4 + 255) / 256, 256, 0, stream>>>(x, x_bf, TOK * DIM);
  cvt_f32_bf16<<<(DIM * WQCOLS / 4 + 255) / 256, 256, 0, stream>>>(wq, wq_bf, DIM * WQCOLS);
  cvt_f32_bf16<<<(HEADS * NKEYS * 2 * DKEY / 4 + 255) / 256, 256, 0, stream>>>(
      ky, ky_bf, HEADS * NKEYS * 2 * DKEY);

  gemm_w2<<<dim3(DIM / 64, NKEYS / 64, 16), 128, 0, stream>>>(wq_bf, ky_bf, w2t);
  gemm_sim<<<dim3(TOK / 64, NCOL / 64), 128, 0, stream>>>(x_bf, w2t, sim);
  argmax_kernel<<<TOK, 256, 0, stream>>>(sim, eidx, esc);
  peer_out_kernel<<<TOK, 256, 0, stream>>>(x, wd, wu, eidx, esc, out);
}
